// seqAutoencoder_24927990186107
// MI455X (gfx1250) — compile-verified, matrix operand order checked by reading, not run on hardware
//
#include <hip/hip_runtime.h>
#include <stdint.h>

#define HIDDEN  512
#define VOCAB   32
#define NSEQ    256
#define SEQLEN  512
#define G3      (3 * HIDDEN)      // 1536 (r,z,n gates)
#define NTILES  (G3 / 16)         // 96 N-tiles of 16
#define KBLKS   (HIDDEN / 32)     // 16 K-blocks of 32
#define MTILE   16
#define NWG     (NSEQ / MTILE)    // 16 persistent workgroups

typedef __attribute__((ext_vector_type(16))) __bf16        bf16x16;
typedef __attribute__((ext_vector_type(8)))  float         floatx8;
typedef __attribute__((ext_vector_type(4)))  unsigned int  u32x4;

union ABreg { bf16x16 v; u32x4 q[2]; };

__device__ __forceinline__ unsigned short f2bf(float f) {
  unsigned int u = __float_as_uint(f);
  return (unsigned short)((u + 0x7FFFu + ((u >> 16) & 1u)) >> 16);
}

// ---------------------------------------------------------------------------
// Phase 0a: gx_table[v][n] = (relu?)(emb[v]) . W_ih[n] + b_ih[n]
// VOCAB=32 collapses the whole input-side GEMM of both GRUs into a table.
// ---------------------------------------------------------------------------
__global__ void gx_table_kernel(const float* __restrict__ emb,
                                const float* __restrict__ W_ih,
                                const float* __restrict__ b_ih,
                                float* __restrict__ gx, int do_relu) {
  int idx = blockIdx.x * blockDim.x + threadIdx.x;   // [0, VOCAB*G3)
  int v = idx / G3;
  int n = idx - v * G3;
  const float* e = emb + v * HIDDEN;
  const float* w = W_ih + n * HIDDEN;
  float acc = b_ih[n];
#pragma unroll 8
  for (int k = 0; k < HIDDEN; ++k) {
    float x = e[k];
    if (do_relu) x = fmaxf(x, 0.f);
    acc = fmaf(x, w[k], acc);
  }
  gx[idx] = acc;
}

// ---------------------------------------------------------------------------
// Phase 0b: pack W (row-major [N,512] f32) into bf16 WMMA B-operand layout:
//   idx = ((ntile*KBLKS + kb)*32 + lane)*16 + j
//   lane: N-col = lane&15 ; K = kb*32 + (j<8?0:16) + (lane>>4)*8 + (j&7)
// Matches ISA 7.12.2 16-bit operand striping so the hot loop is pure b128.
// ---------------------------------------------------------------------------
__global__ void pack_w_kernel(const float* __restrict__ W,
                              unsigned short* __restrict__ dst, int total) {
  int idx = blockIdx.x * blockDim.x + threadIdx.x;
  if (idx >= total) return;
  int j    = idx & 15;
  int lane = (idx >> 4) & 31;
  int kb   = (idx >> 9) & 15;
  int nt   = idx >> 13;
  int n = nt * 16 + (lane & 15);
  int k = kb * 32 + ((j < 8) ? 0 : 16) + ((lane >> 4) << 3) + (j & 7);
  dst[idx] = f2bf(W[n * HIDDEN + k]);
}

// ---------------------------------------------------------------------------
// Persistent GRU scan. One WG = 16 sequences for all 512 steps.
// Per step: gh = h @ W_hh^T via v_wmma_f32_16x16x32_bf16 (12 N-tiles/wave),
// then gate elementwise from the gx lookup table. Decoder additionally does
// a 2-tile WMMA logits GEMM + log-softmax NLL.
// Weight stream (1.5 MB/step >> WGP$) is loaded non-temporal: keep it from
// thrashing the near cache; it stays L2-resident (192 MB) for all 16 WGs.
// ---------------------------------------------------------------------------
template <int DEC>
__global__ __launch_bounds__(256, 1)
void gru_kernel(const int*   __restrict__ tokens,
                const float* __restrict__ gx,        // [VOCAB, G3] L2-hot
                const unsigned short* __restrict__ pW,   // packed W_hh bf16
                const float* __restrict__ b_hh,
                const float* __restrict__ h_init,    // dec: [NSEQ,H]; enc: null
                float*       __restrict__ h_final,   // enc: [NSEQ,H]
                const unsigned short* __restrict__ pWout,
                const float* __restrict__ b_out,
                float*       __restrict__ loss_partial) {
  __shared__ unsigned short h_bf[MTILE][HIDDEN + 16];  // WMMA A operand
  __shared__ float          h_f32[MTILE][HIDDEN + 8];  // full-precision state
  __shared__ float          ghS[MTILE][G3 + 8];        // staged gate pre-acts
  __shared__ float          logitsS[MTILE][VOCAB + 4];
  __shared__ float          redS[MTILE];

  const int tid   = threadIdx.x;
  const int lane  = tid & 31;          // wave32
  const int wave  = tid >> 5;          // 8 waves
  const int lidx  = lane & 15;         // A-row / B,C,D-column within tile
  const int khalf = (lane >> 4) << 3;  // K sub-offset 0|8 (A striping)
  const int mhi   = (lane >> 4) << 3;  // D: M offset 0|8 (C/D striping)
  const int seq0  = blockIdx.x * MTILE;

  for (int e = tid; e < MTILE * HIDDEN; e += 256) {
    int m = e >> 9, j = e & (HIDDEN - 1);
    float hv = DEC ? h_init[(seq0 + m) * HIDDEN + j] : 0.f;
    h_f32[m][j] = hv;
    h_bf[m][j]  = f2bf(hv);
  }
  float nll_acc = 0.f;
  __syncthreads();

  for (int t = 0; t < SEQLEN; ++t) {
    // ---- gh[16,1536] = h[16,512] @ W_hh^T + b_hh  (WMMA bf16) ----
    for (int nt = wave; nt < NTILES; nt += 8) {
      const u32x4* bptr = (const u32x4*)pW + ((size_t)(nt * KBLKS) * 32 + lane) * 2;
      __builtin_prefetch(bptr, 0, 1);  // global_prefetch_b8 on weight stream
      floatx8 acc = {0.f, 0.f, 0.f, 0.f, 0.f, 0.f, 0.f, 0.f};
#pragma unroll
      for (int kb = 0; kb < KBLKS; ++kb) {
        ABreg a, b;
        const unsigned short* hp = &h_bf[lidx][kb * 32 + khalf];
        a.q[0] = *(const u32x4*)hp;
        a.q[1] = *(const u32x4*)(hp + 16);
        b.q[0] = __builtin_nontemporal_load(bptr);      // TH_NT weight stream
        b.q[1] = __builtin_nontemporal_load(bptr + 1);
        bptr += 64;  // next K-block: 32 lanes * 16 bf16
        acc = __builtin_amdgcn_wmma_f32_16x16x32_bf16(
            false, a.v, false, b.v, (short)0, acc, false, false);
      }
      int col  = nt * 16 + lidx;
      float bb = b_hh[col];
#pragma unroll
      for (int r = 0; r < 8; ++r)      // D layout: M = r + (lane>=16)*8
        ghS[mhi + r][col] = acc[r] + bb;
    }
    __syncthreads();

    // ---- gates: r,z sigmoid; n tanh; h' = (1-z)n + z h ----
    for (int e = tid; e < MTILE * HIDDEN; e += 256) {
      int m = e >> 9, j = e & (HIDDEN - 1);
      int tok;
      if (DEC) tok = (t == 0) ? 0 : tokens[(seq0 + m) * SEQLEN + (t - 1)];
      else     tok = tokens[(seq0 + m) * SEQLEN + t];
      const float* g = gx + tok * G3;  // gx table lookup (L2/WGP$-hot)
      float r = 1.f / (1.f + __expf(-(g[j] + ghS[m][j])));
      float z = 1.f / (1.f + __expf(-(g[HIDDEN + j] + ghS[m][HIDDEN + j])));
      float n = tanhf(g[2 * HIDDEN + j] + r * ghS[m][2 * HIDDEN + j]);
      float hv = (1.f - z) * n + z * h_f32[m][j];
      h_f32[m][j] = hv;
      h_bf[m][j]  = f2bf(hv);
    }
    __syncthreads();

    if (DEC) {
      // ---- logits[16,32] = h_new @ W_out^T + b_out (2 WMMA tiles) ----
      if (wave < 2) {
        const u32x4* bptr =
            (const u32x4*)pWout + ((size_t)(wave * KBLKS) * 32 + lane) * 2;
        floatx8 acc = {0.f, 0.f, 0.f, 0.f, 0.f, 0.f, 0.f, 0.f};
#pragma unroll
        for (int kb = 0; kb < KBLKS; ++kb) {
          ABreg a, b;
          const unsigned short* hp = &h_bf[lidx][kb * 32 + khalf];
          a.q[0] = *(const u32x4*)hp;
          a.q[1] = *(const u32x4*)(hp + 16);
          b.q[0] = bptr[0];   // W_out is tiny (32 KB): keep it cached near
          b.q[1] = bptr[1];
          bptr += 64;
          acc = __builtin_amdgcn_wmma_f32_16x16x32_bf16(
              false, a.v, false, b.v, (short)0, acc, false, false);
        }
        int col  = wave * 16 + lidx;
        float bb = b_out[col];
#pragma unroll
        for (int r = 0; r < 8; ++r) logitsS[mhi + r][col] = acc[r] + bb;
      }
      __syncthreads();
      if (tid < MTILE) {               // 32-wide log-softmax + NLL per seq
        int m   = tid;
        int tgt = tokens[(seq0 + m) * SEQLEN + t];
        float mx = logitsS[m][0];
#pragma unroll
        for (int v = 1; v < VOCAB; ++v) mx = fmaxf(mx, logitsS[m][v]);
        float s = 0.f;
#pragma unroll
        for (int v = 0; v < VOCAB; ++v) s += __expf(logitsS[m][v] - mx);
        nll_acc += (mx + __logf(s)) - logitsS[m][tgt];
      }
      __syncthreads();
    }
  }

  if (!DEC) {
    for (int e = tid; e < MTILE * HIDDEN; e += 256) {
      int m = e >> 9, j = e & (HIDDEN - 1);
      h_final[(seq0 + m) * HIDDEN + j] = h_f32[m][j];
    }
  } else {
    if (tid < MTILE) redS[tid] = nll_acc;
    __syncthreads();
    if (tid == 0) {                    // deterministic per-WG partial
      float s = 0.f;
#pragma unroll
      for (int i = 0; i < MTILE; ++i) s += redS[i];
      loss_partial[blockIdx.x] = s;
    }
  }
}

// ---------------------------------------------------------------------------
// embeds = column-mean of h_enc; loss = sum(partials)/(N*L)
// ---------------------------------------------------------------------------
__global__ void finalize_kernel(const float* __restrict__ h_enc,
                                const float* __restrict__ loss_partial,
                                float* __restrict__ out) {
  int j = threadIdx.x;  // 512 threads
  float s = 0.f;
  for (int m = 0; m < NSEQ; ++m) s += h_enc[m * HIDDEN + j];
  out[j] = s * (1.f / (float)NSEQ);
  if (j == 0) {
    float L = 0.f;
#pragma unroll
    for (int i = 0; i < NWG; ++i) L += loss_partial[i];
    out[HIDDEN] = L / (float)(NSEQ * SEQLEN);
  }
}

extern "C" void kernel_launch(void* const* d_in, const int* in_sizes, int n_in,
                              void* d_out, int out_size, void* d_ws, size_t ws_size,
                              hipStream_t stream) {
  (void)in_sizes; (void)n_in; (void)out_size; (void)ws_size;
  const int*   tokens   = (const int*)  d_in[0];
  const float* emb_enc  = (const float*)d_in[1];
  const float* W_ih_enc = (const float*)d_in[2];
  const float* W_hh_enc = (const float*)d_in[3];
  const float* b_ih_enc = (const float*)d_in[4];
  const float* b_hh_enc = (const float*)d_in[5];
  const float* emb_dec  = (const float*)d_in[6];
  const float* W_ih_dec = (const float*)d_in[7];
  const float* W_hh_dec = (const float*)d_in[8];
  const float* b_ih_dec = (const float*)d_in[9];
  const float* b_hh_dec = (const float*)d_in[10];
  const float* W_out    = (const float*)d_in[11];
  const float* b_out    = (const float*)d_in[12];
  float* out = (float*)d_out;

  char* ws = (char*)d_ws;
  size_t off = 0;
  auto alloc = [&](size_t bytes) -> void* {
    void* p = ws + off;
    off = (off + bytes + 255) & ~(size_t)255;
    return p;
  };
  float*          gx_enc  = (float*)alloc((size_t)VOCAB * G3 * 4);
  float*          gx_dec  = (float*)alloc((size_t)VOCAB * G3 * 4);
  unsigned short* pW_enc  = (unsigned short*)alloc((size_t)G3 * HIDDEN * 2);
  unsigned short* pW_dec  = (unsigned short*)alloc((size_t)G3 * HIDDEN * 2);
  unsigned short* pW_out  = (unsigned short*)alloc((size_t)VOCAB * HIDDEN * 2);
  float*          h_enc   = (float*)alloc((size_t)NSEQ * HIDDEN * 4);
  float*          lossP   = (float*)alloc((size_t)NWG * 4);

  dim3 b256(256);
  gx_table_kernel<<<(VOCAB * G3) / 256, b256, 0, stream>>>(
      emb_enc, W_ih_enc, b_ih_enc, gx_enc, 0);
  gx_table_kernel<<<(VOCAB * G3) / 256, b256, 0, stream>>>(
      emb_dec, W_ih_dec, b_ih_dec, gx_dec, 1);
  pack_w_kernel<<<(G3 * HIDDEN) / 256, b256, 0, stream>>>(
      W_hh_enc, pW_enc, G3 * HIDDEN);
  pack_w_kernel<<<(G3 * HIDDEN) / 256, b256, 0, stream>>>(
      W_hh_dec, pW_dec, G3 * HIDDEN);
  pack_w_kernel<<<(VOCAB * HIDDEN) / 256, b256, 0, stream>>>(
      W_out, pW_out, VOCAB * HIDDEN);

  gru_kernel<0><<<NWG, b256, 0, stream>>>(
      tokens, gx_enc, pW_enc, b_hh_enc, nullptr, h_enc, nullptr, nullptr, nullptr);
  gru_kernel<1><<<NWG, b256, 0, stream>>>(
      tokens, gx_dec, pW_dec, b_hh_dec, h_enc, nullptr, pW_out, b_out, lossP);

  finalize_kernel<<<1, 512, 0, stream>>>(h_enc, lossP, out);
}
